// MHA_86440511799819
// MI455X (gfx1250) — matrix operand order repaired and checked by reading, not compiled, not run on hardware
//
#include <hip/hip_runtime.h>
#include <hip/hip_bf16.h>

// ---------------------------------------------------------------------------
// MHA forward for MI455X (gfx1250, wave32, WMMA + Tensor Data Mover).
// Pipeline: cvt(f32->f16) -> QKV GEMM+bias+RoPE (WMMA, TDM double-buffered)
// -> flash attention (WMMA QK^T + PV, online softmax) -> out GEMM+bias -> f32.
// ---------------------------------------------------------------------------

typedef __attribute__((ext_vector_type(16))) _Float16 v16h;
typedef __attribute__((ext_vector_type(8)))  _Float16 v8h;
typedef __attribute__((ext_vector_type(4)))  _Float16 v4h;
typedef __attribute__((ext_vector_type(8)))  float    v8f;
typedef __attribute__((ext_vector_type(4)))  unsigned int v4u;
typedef __attribute__((ext_vector_type(8)))  int      v8i;
typedef __attribute__((ext_vector_type(4)))  int      v4i;

#define NHEADS   16
#define HDIM     128
#define SEQ      2048
#define BATCH    4
#define EMB      2048          // NHEADS*HDIM
#define MTOT     (BATCH*SEQ)   // 8192
#define KDIM     2048          // reduction dim for both GEMMs
#define LDT      40            // LDS tile stride (32 + 8 pad) in halves
#define KTS      136           // attention K-tile LDS stride in halves
#define TILE_H   (128*LDT)     // halves per GEMM LDS tile buffer
#define ROPE_LN  0.14391156f   // ln(10000)/64

static __device__ __forceinline__ v16h frag_cat(const _Float16* p0, const _Float16* p1) {
    v8h lo = *reinterpret_cast<const v8h*>(p0);
    v8h hi = *reinterpret_cast<const v8h*>(p1);
    return __builtin_shufflevector(lo, hi, 0,1,2,3,4,5,6,7,8,9,10,11,12,13,14,15);
}
static __device__ __forceinline__ v16h frag_16(const _Float16* p) {
    return frag_cat(p, p + 8);
}

// ---------------------------------------------------------------------------
// TDM descriptor helpers (CDNA5 D#, CDNA5 ISA).
// Loads a 128-row x 32-half tile (row stride KDIM halves in memory) into LDS
// with +4 DWORD padding every 16 DWORDs => LDS row stride = 40 halves (LDT).
// ---------------------------------------------------------------------------
static __device__ __forceinline__ unsigned lds_byte_off(const void* p) {
    // LDS aperture: flat addr[31:0] is the LDS byte address (ISA 10.2).
    return (unsigned)(unsigned long long)(uintptr_t)p;
}
static __device__ __forceinline__ v4u tdm_group0(unsigned lds_off, const void* gaddr) {
    unsigned long long ga = (unsigned long long)(uintptr_t)gaddr;
    v4u g0;
    g0[0] = 1u;                                   // count=1, user descriptor
    g0[1] = lds_off;                              // lds_addr
    g0[2] = (unsigned)ga;                         // global_addr[31:0]
    g0[3] = (unsigned)(ga >> 32) | (2u << 30);    // global_addr[56:32] | type=2
    return g0;
}
static __device__ __forceinline__ v8i tdm_group1_gemm() {
    v8i g1;
    // data_size=1 (2B), pad_enable, pad_interval=3 (16 DWORDs), pad_amount=3 (4 DWORDs)
    g1[0] = (1 << 16) | (1 << 20) | (3 << 22) | (3 << 25);
    const int td0 = KDIM;            // tensor_dim0 (elements, contiguous)
    const int td1 = 0x100000;        // tensor_dim1 (rows; generous, no OOB)
    g1[1] = (td0 & 0xFFFF) << 16;                       // bits 79:48 start
    g1[2] = (td0 >> 16) | ((td1 & 0xFFFF) << 16);       // tensor_dim0 hi | dim1 lo
    g1[3] = (td1 >> 16) | (32 << 16);                   // dim1 hi | tile_dim0=32
    g1[4] = 128;                                        // tile_dim1=128, tile_dim2=0
    g1[5] = KDIM;                                       // tensor_dim0_stride[31:0]
    g1[6] = 0;                                          // stride hi | dim1_stride lo
    g1[7] = 0;
    return g1;
}
static __device__ __forceinline__ void tdm_load(v4u g0, v8i g1) {
    v4i z4; z4[0] = 0; z4[1] = 0; z4[2] = 0; z4[3] = 0;
    v8i z8; z8[0] = 0; z8[1] = 0; z8[2] = 0; z8[3] = 0;
            z8[4] = 0; z8[5] = 0; z8[6] = 0; z8[7] = 0;
    __builtin_amdgcn_tensor_load_to_lds(g0, g1, z4, z4, z8, 0);
}

// ---------------------------------------------------------------------------
// f32 -> f16 conversion (4-wide)
// ---------------------------------------------------------------------------
__global__ void cvt_f32_f16_kernel(const float4* __restrict__ src,
                                   _Float16* __restrict__ dst, int n4) {
    int i = blockIdx.x * blockDim.x + threadIdx.x;
    if (i >= n4) return;
    float4 v = src[i];
    v4h h; h[0] = (_Float16)v.x; h[1] = (_Float16)v.y;
           h[2] = (_Float16)v.z; h[3] = (_Float16)v.w;
    *reinterpret_cast<v4h*>(dst + i * 4) = h;
}

// ---------------------------------------------------------------------------
// WMMA GEMM mainloop with TDM double buffering:
// C[128x128] += A[128xK] * W[128xK]^T.  256 threads = 8 waves; wave w owns
// rows [w*16, w*16+16) x all 128 cols (8 accumulators). Wave 0 drives the
// Tensor Data Mover: next K-tile DMA overlaps current-tile WMMAs.
// ---------------------------------------------------------------------------
static __device__ __forceinline__ void gemm128_mainloop(
        const _Float16* __restrict__ A, const _Float16* __restrict__ Bw,
        int mbase, int nbase, _Float16* As, _Float16* Ws, v8f* acc) {
    const int tid  = threadIdx.x;
    const int lane = tid & 31;
    const int w    = tid >> 5;
    const int ln   = lane & 15;
    const int hi   = lane >> 4;      // lane half: selects K sub-range
    const int kb   = hi * 8;

    const _Float16* aTile = A  + (size_t)mbase * KDIM;
    const _Float16* bTile = Bw + (size_t)nbase * KDIM;

    const v8i g1 = tdm_group1_gemm();
    const unsigned asOff0 = lds_byte_off(As);
    const unsigned wsOff0 = lds_byte_off(Ws);

    if (w == 0) {   // prologue: DMA first K-tile into buffer 0
        tdm_load(tdm_group0(asOff0, aTile), g1);
        tdm_load(tdm_group0(wsOff0, bTile), g1);
    }

    for (int k0 = 0; k0 < KDIM; k0 += 32) {
        const int buf = (k0 >> 5) & 1;
        if (w == 0) {
            if (k0 + 32 < KDIM) {   // DMA next tile into the other buffer
                const unsigned boff = (buf ^ 1) * (TILE_H * 2);
                tdm_load(tdm_group0(asOff0 + boff, aTile + k0 + 32), g1);
                tdm_load(tdm_group0(wsOff0 + boff, bTile + k0 + 32), g1);
                __builtin_amdgcn_s_wait_tensorcnt((short)2);  // current buf done
            } else {
                __builtin_amdgcn_s_wait_tensorcnt((short)0);
            }
        }
        __syncthreads();   // publish TDM-written tile to all waves

        const _Float16* AsB = As + buf * TILE_H;
        const _Float16* WsB = Ws + buf * TILE_H;

        // A fragment: lane holds row (w*16+ln), K = hi*16+i
        const _Float16* arow = &AsB[(w * 16 + ln) * LDT];
        v16h afrag = frag_16(arow + hi * 16);
        // B fragments first (lets LDS reads overlap WMMA issue)
        v16h bfr[8];
        #pragma unroll
        for (int nt = 0; nt < 8; ++nt)
            bfr[nt] = frag_16(&WsB[(nt * 16 + ln) * LDT + hi * 16]);
        #pragma unroll
        for (int nt = 0; nt < 8; ++nt)
            acc[nt] = __builtin_amdgcn_wmma_f32_16x16x32_f16(
                false, afrag, false, bfr[nt], (short)0, acc[nt], false, false);

        __syncthreads();   // all reads done before TDM overwrites this buffer
    }
}

// ---------------------------------------------------------------------------
// QKV GEMM + bias + RoPE, writing q/k/v in [B,H,S,D] f16.
// grid = (MTOT/128, 48); column block nblk covers exactly one (q|k|v, head).
// RoPE pair (d, d+64) lives in acc[nt] / acc[nt+4] of the same lane/element.
// ---------------------------------------------------------------------------
__global__ __launch_bounds__(256) void qkv_rope_gemm_kernel(
        const _Float16* __restrict__ xh, const _Float16* __restrict__ wh,
        const float* __restrict__ bias,
        _Float16* __restrict__ qh, _Float16* __restrict__ kh,
        _Float16* __restrict__ vh) {
    __shared__ __align__(16) _Float16 As[2 * TILE_H];
    __shared__ __align__(16) _Float16 Ws[2 * TILE_H];

    v8f acc[8];
    #pragma unroll
    for (int i = 0; i < 8; ++i)
        #pragma unroll
        for (int j = 0; j < 8; ++j) acc[i][j] = 0.0f;

    const int mbase = blockIdx.x * 128;
    const int nblk  = blockIdx.y;
    const int nbase = nblk * 128;
    gemm128_mainloop(xh, wh, mbase, nbase, As, Ws, acc);

    const int i3 = nblk >> 4;          // 0=q, 1=k, 2=v
    const int h  = nblk & 15;
    const int tid = threadIdx.x, lane = tid & 31;
    const int w = tid >> 5, ln = lane & 15, hi = lane >> 4;
    _Float16* base = (i3 == 0 ? qh : (i3 == 1 ? kh : vh));

    #pragma unroll
    for (int e = 0; e < 8; ++e) {
        const int grow = mbase + w * 16 + e + hi * 8;   // C layout: M=e (+8 hi-half)
        const int b = grow >> 11;                        // / SEQ
        const int s = grow & (SEQ - 1);
        _Float16* dst = base + ((size_t)(b * NHEADS + h) * SEQ + s) * HDIM;
        #pragma unroll
        for (int nt = 0; nt < 4; ++nt) {
            const int d = nt * 16 + ln;                  // 0..63
            float x1 = acc[nt][e]     + bias[nbase + d];
            float x2 = acc[nt + 4][e] + bias[nbase + d + 64];
            if (i3 < 2) {
                float ang = (float)s * __expf(-(float)d * ROPE_LN);
                float sv, cv; __sincosf(ang, &sv, &cv);
                dst[d]      = (_Float16)(x1 * cv - x2 * sv);
                dst[d + 64] = (_Float16)(x2 * cv + x1 * sv);
            } else {
                dst[d]      = (_Float16)x1;
                dst[d + 64] = (_Float16)x2;
            }
        }
    }
}

// ---------------------------------------------------------------------------
// Flash attention: grid = (SEQ/64, B*H), 128 threads = 4 waves.
// Each wave owns 16 query rows; key blocks of 32 staged in LDS.
// QK^T: 4 chained WMMA per 16x16 score tile; PV: 8 WMMA over D=128.
// ---------------------------------------------------------------------------
__global__ __launch_bounds__(128) void flash_attn_kernel(
        const _Float16* __restrict__ qh, const _Float16* __restrict__ kh,
        const _Float16* __restrict__ vh, _Float16* __restrict__ ctxh) {
    __shared__ __align__(16) _Float16 Kt[32 * KTS];        // [key][k]
    __shared__ __align__(16) _Float16 VtT[128 * LDT];      // [d][key] (transposed)
    __shared__ __align__(16) _Float16 Pst[4 * 16 * 32];    // per-wave P stage

    const int qb = blockIdx.x, bh = blockIdx.y;
    const int b = bh >> 4, h = bh & 15;
    const _Float16* Q = qh + (size_t)bh * SEQ * HDIM;
    const _Float16* K = kh + (size_t)bh * SEQ * HDIM;
    const _Float16* V = vh + (size_t)bh * SEQ * HDIM;

    const int tid = threadIdx.x, lane = tid & 31;
    const int w = tid >> 5, ln = lane & 15, hi = lane >> 4;
    const int kb = hi * 8;
    const int qrow0 = qb * 64 + w * 16;

    // preload Q fragments for all four 32-wide K chunks of D=128
    v16h qa[4];
    const _Float16* qrow = Q + (size_t)(qrow0 + ln) * HDIM;
    #pragma unroll
    for (int c = 0; c < 4; ++c)
        qa[c] = frag_16(qrow + c * 32 + hi * 16);

    v8f acc[8];
    float mstat[8], lstat[8];
    #pragma unroll
    for (int i = 0; i < 8; ++i) {
        mstat[i] = -3.0e38f; lstat[i] = 0.0f;
        #pragma unroll
        for (int j = 0; j < 8; ++j) acc[i][j] = 0.0f;
    }

    const float scale = 0.08838834764831845f;   // 1/sqrt(128)
    const int kend = qb * 64 + 64;

    for (int kb0 = 0; kb0 < kend; kb0 += 32) {
        // --- stage K (row-major) and V (transposed) into LDS ---
        #pragma unroll
        for (int i = 0; i < 2; ++i) {                   // 256 chunks of 16 halves
            const int c = tid * 2 + i;
            const int r = c >> 3, col = (c & 7) << 4;
            const _Float16* src = K + (size_t)(kb0 + r) * HDIM + col;
            *reinterpret_cast<v8h*>(&Kt[r * KTS + col])     = *reinterpret_cast<const v8h*>(src);
            *reinterpret_cast<v8h*>(&Kt[r * KTS + col + 8]) = *reinterpret_cast<const v8h*>(src + 8);
        }
        #pragma unroll
        for (int j = 0; j < 4; ++j) {                   // 512 chunks of 8 halves
            const int c = tid * 4 + j;
            const int r = c >> 4, col = (c & 15) << 3;
            v8h vv = *reinterpret_cast<const v8h*>(V + (size_t)(kb0 + r) * HDIM + col);
            #pragma unroll
            for (int t = 0; t < 8; ++t) VtT[(col + t) * LDT + r] = vv[t];
        }
        __syncthreads();

        // --- scores: S = Q K^T (16x32 per wave) ---
        v8f sc[2];
        #pragma unroll
        for (int nt = 0; nt < 2; ++nt) {
            #pragma unroll
            for (int j = 0; j < 8; ++j) sc[nt][j] = 0.0f;
            #pragma unroll
            for (int c = 0; c < 4; ++c) {
                v16h bfrag = frag_16(&Kt[(nt * 16 + ln) * KTS + c * 32 + hi * 16]);
                sc[nt] = __builtin_amdgcn_wmma_f32_16x16x32_f16(
                    false, qa[c], false, bfrag, (short)0, sc[nt], false, false);
            }
        }

        // --- online softmax (row stats live across 16-lane half-groups) ---
        float p0v[8], p1v[8];
        #pragma unroll
        for (int e = 0; e < 8; ++e) {
            const int row = qrow0 + e + hi * 8;
            float s0 = sc[0][e] * scale;
            float s1 = sc[1][e] * scale;
            if (kb0 + ln      > row) s0 = -3.0e38f;
            if (kb0 + 16 + ln > row) s1 = -3.0e38f;
            float mx = fmaxf(s0, s1);
            #pragma unroll
            for (int off = 1; off < 16; off <<= 1)
                mx = fmaxf(mx, __shfl_xor(mx, off, 32));
            const float mnew = fmaxf(mstat[e], mx);
            const float p0 = __expf(s0 - mnew);
            const float p1 = __expf(s1 - mnew);
            float rs = p0 + p1;
            #pragma unroll
            for (int off = 1; off < 16; off <<= 1)
                rs += __shfl_xor(rs, off, 32);
            const float corr = __expf(mstat[e] - mnew);
            lstat[e] = lstat[e] * corr + rs;
            mstat[e] = mnew;
            #pragma unroll
            for (int dt = 0; dt < 8; ++dt) acc[dt][e] *= corr;
            p0v[e] = p0; p1v[e] = p1;
        }

        // --- relayout P (C-layout f32 -> A-layout f16) through LDS ---
        _Float16* pw = &Pst[w * 16 * 32];
        #pragma unroll
        for (int e = 0; e < 8; ++e) {
            const int row = e + hi * 8;
            pw[row * 32 + ln]      = (_Float16)p0v[e];
            pw[row * 32 + 16 + ln] = (_Float16)p1v[e];
        }
        __syncthreads();

        // --- PV: acc += P (16x32) * V (32x128) ---
        v16h pa = frag_16(&pw[ln * 32 + hi * 16]);
        #pragma unroll
        for (int dt = 0; dt < 8; ++dt) {
            v16h vb = frag_16(&VtT[(dt * 16 + ln) * LDT + hi * 16]);
            acc[dt] = __builtin_amdgcn_wmma_f32_16x16x32_f16(
                false, pa, false, vb, (short)0, acc[dt], false, false);
        }
        __syncthreads();
    }

    // --- normalize + write ctx [B,S,E] f16 ---
    #pragma unroll
    for (int e = 0; e < 8; ++e) {
        const float rl = 1.0f / lstat[e];
        const int srow = qrow0 + e + hi * 8;
        _Float16* dst = ctxh + ((size_t)(b * SEQ + srow)) * EMB + h * HDIM;
        #pragma unroll
        for (int dt = 0; dt < 8; ++dt)
            dst[dt * 16 + ln] = (_Float16)(acc[dt][e] * rl);
    }
}

// ---------------------------------------------------------------------------
// Output GEMM + bias -> f32. grid = (MTOT/128, EMB/128)
// ---------------------------------------------------------------------------
__global__ __launch_bounds__(256) void out_gemm_kernel(
        const _Float16* __restrict__ ctxh, const _Float16* __restrict__ wh,
        const float* __restrict__ bias, float* __restrict__ out) {
    __shared__ __align__(16) _Float16 As[2 * TILE_H];
    __shared__ __align__(16) _Float16 Ws[2 * TILE_H];

    v8f acc[8];
    #pragma unroll
    for (int i = 0; i < 8; ++i)
        #pragma unroll
        for (int j = 0; j < 8; ++j) acc[i][j] = 0.0f;

    const int mbase = blockIdx.x * 128;
    const int nbase = blockIdx.y * 128;
    gemm128_mainloop(ctxh, wh, mbase, nbase, As, Ws, acc);

    const int tid = threadIdx.x, lane = tid & 31;
    const int w = tid >> 5, ln = lane & 15, hi = lane >> 4;
    #pragma unroll
    for (int e = 0; e < 8; ++e) {
        const int grow = mbase + w * 16 + e + hi * 8;
        #pragma unroll
        for (int nt = 0; nt < 8; ++nt) {
            const int col = nbase + nt * 16 + ln;
            out[(size_t)grow * EMB + col] = acc[nt][e] + bias[col];
        }
    }
}

// ---------------------------------------------------------------------------
// Host-side launch
// ---------------------------------------------------------------------------
extern "C" void kernel_launch(void* const* d_in, const int* in_sizes, int n_in,
                              void* d_out, int out_size, void* d_ws, size_t ws_size,
                              hipStream_t stream) {
    const float* x      = (const float*)d_in[0];   // [B,S,E]
    const float* WqkvW  = (const float*)d_in[1];   // [3E,E]
    const float* WqkvB  = (const float*)d_in[2];   // [3E]
    const float* outW   = (const float*)d_in[3];   // [E,E]
    const float* outB   = (const float*)d_in[4];   // [E]
    float* out = (float*)d_out;

    char* ws = (char*)d_ws;
    size_t off = 0;
    _Float16* xh    = (_Float16*)(ws + off); off += (size_t)MTOT * EMB * 2;      // 32 MB
    _Float16* wqkvh = (_Float16*)(ws + off); off += (size_t)3 * EMB * EMB * 2;   // 24 MB
    _Float16* outwh = (_Float16*)(ws + off); off += (size_t)EMB * EMB * 2;       //  8 MB
    _Float16* qh    = (_Float16*)(ws + off); off += (size_t)MTOT * EMB * 2;      // 32 MB
    _Float16* kh    = (_Float16*)(ws + off); off += (size_t)MTOT * EMB * 2;      // 32 MB
    _Float16* vh    = (_Float16*)(ws + off); off += (size_t)MTOT * EMB * 2;      // 32 MB
    _Float16* ctxh  = (_Float16*)(ws + off); off += (size_t)MTOT * EMB * 2;      // 32 MB

    // 1) precision conversion
    {
        int n4 = (MTOT * EMB) / 4;
        cvt_f32_f16_kernel<<<(n4 + 255) / 256, 256, 0, stream>>>((const float4*)x, xh, n4);
        n4 = (3 * EMB * EMB) / 4;
        cvt_f32_f16_kernel<<<(n4 + 255) / 256, 256, 0, stream>>>((const float4*)WqkvW, wqkvh, n4);
        n4 = (EMB * EMB) / 4;
        cvt_f32_f16_kernel<<<(n4 + 255) / 256, 256, 0, stream>>>((const float4*)outW, outwh, n4);
    }

    // 2) QKV projection + bias + RoPE  (N = 3E = 48 column blocks of 128)
    {
        dim3 grid(MTOT / 128, 48);
        qkv_rope_gemm_kernel<<<grid, 256, 0, stream>>>(xh, wqkvh, WqkvB, qh, kh, vh);
    }

    // 3) causal flash attention
    {
        dim3 grid(SEQ / 64, BATCH * NHEADS);
        flash_attn_kernel<<<grid, 128, 0, stream>>>(qh, kh, vh, ctxh);
    }

    // 4) output projection + bias
    {
        dim3 grid(MTOT / 128, EMB / 128);
        out_gemm_kernel<<<grid, 256, 0, stream>>>(ctxh, outwh, outB, out);
    }
}
